// SingleHead_4664334483563
// MI455X (gfx1250) — matrix operand order repaired and checked by reading, not compiled, not run on hardware
//
#include <hip/hip_runtime.h>
#include <hip/hip_bf16.h>
#include <stddef.h>

// ---------------------------------------------------------------------------
// Types for CDNA5 WMMA (gfx1250, wave32)
// ---------------------------------------------------------------------------
typedef __attribute__((ext_vector_type(16))) __bf16 v16bf;
typedef __attribute__((ext_vector_type(8)))  __bf16 v8bf;
typedef __attribute__((ext_vector_type(8)))  float  v8f;

static constexpr int Bn = 8;
static constexpr int Tn = 2048;
static constexpr int En = 1024;
static constexpr int Dh = 64;

// q is pre-scaled by (1/sqrt(D)) * log2(e) so softmax can use exp2 (v_exp_f32)
static constexpr float QSCALE = 0.125f * 1.44269504088896340736f;

__device__ __forceinline__ v8f wmma_bf16(v16bf a, v16bf b, v8f c) {
  // D = A(16x32 bf16) * B(32x16 bf16) + C(16x16 f32)
  return __builtin_amdgcn_wmma_f32_16x16x32_bf16(
      /*neg_a=*/false, a, /*neg_b=*/false, b,
      /*c_mod=*/(short)0, c, /*reuse_a=*/false, /*reuse_b=*/false);
}

__device__ __forceinline__ v16bf pack_bf16(v8f a0, v8f a1) {
  v16bf r;
#pragma unroll
  for (int i = 0; i < 8; ++i) {
    r[i]     = (__bf16)a0[i];
    r[i + 8] = (__bf16)a1[i];
  }
  return r;
}

__device__ __forceinline__ v16bf cat8(v8bf lo, v8bf hi) {
  return __builtin_shufflevector(lo, hi, 0, 1, 2, 3, 4, 5, 6, 7,
                                         8, 9, 10, 11, 12, 13, 14, 15);
}

// ---------------------------------------------------------------------------
// Kernel 0: convert + transpose weights -> Wt[j][d][e] (bf16), j in {q,k,v}
// ---------------------------------------------------------------------------
__global__ void sh_prep_weights(const float* __restrict__ Wq,
                                const float* __restrict__ Wk,
                                const float* __restrict__ Wv,
                                __bf16* __restrict__ Wt) {
  int idx = blockIdx.x * blockDim.x + threadIdx.x;
  const int total = 3 * En * Dh;
  if (idx >= total) return;
  int j = idx / (En * Dh);
  int r = idx % (En * Dh);
  int d = r / En;
  int e = r % En;
  const float* W = (j == 0) ? Wq : ((j == 1) ? Wk : Wv);
  Wt[idx] = (__bf16)W[(size_t)e * Dh + d];
}

// ---------------------------------------------------------------------------
// Kernel 1: QKV projection via WMMA bf16.
//   block = 128 threads (4 waves); wave owns 16 rows of x.
//   Writes q_bf (scaled), k_bf row-major, vT_bf transposed [b][d][t].
// ---------------------------------------------------------------------------
__global__ void __launch_bounds__(128)
sh_qkv_proj(const float* __restrict__ x,
            const float* __restrict__ bq,
            const float* __restrict__ bk,
            const float* __restrict__ bv,
            const __bf16* __restrict__ Wt,
            __bf16* __restrict__ qbf,
            __bf16* __restrict__ kbf,
            __bf16* __restrict__ vT) {
  const int wave = threadIdx.x >> 5;
  const int lane = threadIdx.x & 31;
  const int ln   = lane & 15;
  const bool hi  = lane >= 16;
  const int loK  = hi ? 16 : 0;   // A-frag K offset for this half-wave
  const int bK   = hi ? 16 : 0;   // B-frag K offset for this half-wave

  const size_t r0 = (size_t)blockIdx.x * 64 + (size_t)wave * 16;
  const float* xrow = x + (r0 + ln) * En;

  v8f acc[12];
#pragma unroll
  for (int i = 0; i < 12; ++i) acc[i] = (v8f)0.0f;

  for (int e0 = 0; e0 < En; e0 += 32) {
    // A fragment: 16 rows of x, K-chunk of 32 (convert f32 -> bf16 on the fly)
    v8f a0 = *(const v8f*)&xrow[e0 + loK];
    v8f a1 = *(const v8f*)&xrow[e0 + 8 + loK];
    v16bf af = pack_bf16(a0, a1);
#pragma unroll
    for (int j = 0; j < 3; ++j) {
#pragma unroll
      for (int nt = 0; nt < 4; ++nt) {
        // B fragment: lane = output column, contiguous K from transposed W
        const __bf16* wp =
            Wt + (size_t)(j * Dh + nt * 16 + ln) * En + e0 + bK;
        v16bf bfv = *(const v16bf*)wp;
        acc[j * 4 + nt] = wmma_bf16(af, bfv, acc[j * 4 + nt]);
      }
    }
  }

  // Epilogue: bias add, convert to bf16, scatter q/k row-major and v transposed
#pragma unroll
  for (int nt = 0; nt < 4; ++nt) {
    const int col = nt * 16 + ln;
    const float bqv = bq[col], bkv = bk[col], bvv = bv[col];
#pragma unroll
    for (int r = 0; r < 8; ++r) {
      const size_t row = r0 + r + (hi ? 8 : 0);
      const float qv = (acc[nt][r] + bqv) * QSCALE;
      const float kv = acc[4 + nt][r] + bkv;
      const float vv = acc[8 + nt][r] + bvv;
      qbf[row * Dh + col] = (__bf16)qv;
      kbf[row * Dh + col] = (__bf16)kv;
      const int b = (int)(row >> 11);       // row / T
      const int t = (int)(row & (Tn - 1));  // row % T
      vT[((size_t)(b * Dh + col)) * Tn + t] = (__bf16)vv;
    }
  }
}

// ---------------------------------------------------------------------------
// Kernel 2: per-wave causal flash attention.
//   block = 128 threads (4 waves); wave owns a 16-row query tile.
//   Each iteration handles 32 keys: 4 WMMA for S, online softmax, 4 WMMA for PV.
// ---------------------------------------------------------------------------
__global__ void __launch_bounds__(128)
sh_flash_attn(const __bf16* __restrict__ qbf,
              const __bf16* __restrict__ kbf,
              const __bf16* __restrict__ vT,
              float* __restrict__ out) {
  const int wave = threadIdx.x >> 5;
  const int lane = threadIdx.x & 31;
  const int ln   = lane & 15;
  const bool hi  = lane >= 16;
  const int loK  = hi ? 16 : 0;
  const int bK   = hi ? 16 : 0;

  const int tilesPerBatch = Tn / 64;  // 32
  const int b  = blockIdx.x / tilesPerBatch;
  const int tg = blockIdx.x % tilesPerBatch;
  const int m0 = tg * 64 + wave * 16;  // first query row of this wave (within batch)

  __shared__ __bf16 pbuf[4][16 * 40];  // per-wave P tile, row stride 40 (16B aligned)
  __bf16* pb = pbuf[wave];

  // ---- load Q fragments (D = 64 -> two K=32 chunks), already softmax-scaled
  const __bf16* qrow = qbf + ((size_t)b * Tn + m0 + ln) * Dh;
  v16bf qf[2];
#pragma unroll
  for (int kc = 0; kc < 2; ++kc) {
    v8bf a0 = *(const v8bf*)&qrow[kc * 32 + loK];
    v8bf a1 = *(const v8bf*)&qrow[kc * 32 + 8 + loK];
    qf[kc] = cat8(a0, a1);
  }

  v8f o[4];
#pragma unroll
  for (int i = 0; i < 4; ++i) o[i] = (v8f)0.0f;
  float mrow[8], lrow[8];
#pragma unroll
  for (int r = 0; r < 8; ++r) { mrow[r] = -3.0e38f; lrow[r] = 0.0f; }

  const int sEnd = m0 + 16;  // exclusive key bound (causal)
  for (int s0 = 0; s0 < sEnd; s0 += 32) {
    if (s0 + 32 < sEnd) {
      __builtin_prefetch(&kbf[((size_t)b * Tn + s0 + 32 + ln) * Dh], 0, 0);
    }
    // ---- S = Q @ K^T  (16 x 32 tile as two 16x16 WMMA accumulators)
    v8f s[2];
    s[0] = (v8f)0.0f;
    s[1] = (v8f)0.0f;
#pragma unroll
    for (int nt = 0; nt < 2; ++nt) {
      const __bf16* krow = kbf + ((size_t)b * Tn + s0 + nt * 16 + ln) * Dh;
#pragma unroll
      for (int kc = 0; kc < 2; ++kc) {
        v16bf kf = *(const v16bf*)&krow[kc * 32 + bK];
        s[nt] = wmma_bf16(qf[kc], kf, s[nt]);
      }
    }

    // ---- causal mask + online softmax (exp2 domain; scale folded into q)
#pragma unroll
    for (int r = 0; r < 8; ++r) {
      const int trow = m0 + r + (hi ? 8 : 0);
      float x0 = s[0][r];
      float x1 = s[1][r];
      if (s0 + ln > trow)      x0 = -3.0e38f;
      if (s0 + 16 + ln > trow) x1 = -3.0e38f;

      float mx = fmaxf(x0, x1);
      mx = fmaxf(mx, __shfl_xor(mx, 1, 32));
      mx = fmaxf(mx, __shfl_xor(mx, 2, 32));
      mx = fmaxf(mx, __shfl_xor(mx, 4, 32));
      mx = fmaxf(mx, __shfl_xor(mx, 8, 32));
      const float mnew = fmaxf(mrow[r], mx);
      const float corr = exp2f(mrow[r] - mnew);
      mrow[r] = mnew;

      const float p0 = exp2f(x0 - mnew);
      const float p1 = exp2f(x1 - mnew);
      float rs = p0 + p1;
      rs += __shfl_xor(rs, 1, 32);
      rs += __shfl_xor(rs, 2, 32);
      rs += __shfl_xor(rs, 4, 32);
      rs += __shfl_xor(rs, 8, 32);
      lrow[r] = lrow[r] * corr + rs;

      o[0][r] = o[0][r] * corr;
      o[1][r] = o[1][r] * corr;
      o[2][r] = o[2][r] * corr;
      o[3][r] = o[3][r] * corr;

      // stash P (C-layout -> LDS tile) for A-fragment reload
      const int prow = r + (hi ? 8 : 0);
      pb[prow * 40 + ln]      = (__bf16)p0;
      pb[prow * 40 + 16 + ln] = (__bf16)p1;
    }

    asm volatile("" ::: "memory");  // keep LDS writes before reads (in-order per wave)

    // ---- reload P as A fragment (16 x 32)
    v8bf p0v = *(const v8bf*)&pb[ln * 40 + loK];
    v8bf p1v = *(const v8bf*)&pb[ln * 40 + 8 + loK];
    v16bf pf = cat8(p0v, p1v);

    asm volatile("" ::: "memory");

    // ---- O += P @ V   (B fragment = contiguous keys from transposed V)
#pragma unroll
    for (int dt = 0; dt < 4; ++dt) {
      const __bf16* vp =
          vT + ((size_t)(b * Dh + dt * 16 + ln)) * Tn + s0 + bK;
      v16bf vf = *(const v16bf*)vp;
      o[dt] = wmma_bf16(pf, vf, o[dt]);
    }
  }

  // ---- epilogue: normalize and store f32
#pragma unroll
  for (int r = 0; r < 8; ++r) {
    const float inv = 1.0f / lrow[r];
    const int trow = m0 + r + (hi ? 8 : 0);
    float* orow = out + ((size_t)b * Tn + trow) * Dh;
#pragma unroll
    for (int dt = 0; dt < 4; ++dt) {
      orow[dt * 16 + ln] = o[dt][r] * inv;
    }
  }
}

// ---------------------------------------------------------------------------
// Host launcher
// ---------------------------------------------------------------------------
extern "C" void kernel_launch(void* const* d_in, const int* in_sizes, int n_in,
                              void* d_out, int out_size, void* d_ws, size_t ws_size,
                              hipStream_t stream) {
  (void)in_sizes; (void)n_in; (void)out_size; (void)ws_size;

  const float* x  = (const float*)d_in[0];
  const float* Wq = (const float*)d_in[1];
  const float* bq = (const float*)d_in[2];
  const float* Wk = (const float*)d_in[3];
  const float* bk = (const float*)d_in[4];
  const float* Wv = (const float*)d_in[5];
  const float* bv = (const float*)d_in[6];
  float* out = (float*)d_out;

  // workspace layout (bytes)
  char* ws = (char*)d_ws;
  __bf16* Wt  = (__bf16*)(ws);                               // 3*64*1024*2 = 384 KB
  __bf16* qbf = (__bf16*)(ws + 0x60000);                     // 2 MB
  __bf16* kbf = (__bf16*)(ws + 0x60000 + 0x200000);          // 2 MB
  __bf16* vT  = (__bf16*)(ws + 0x60000 + 0x400000);          // 2 MB

  {
    const int total = 3 * En * Dh;
    const int thr = 256;
    sh_prep_weights<<<(total + thr - 1) / thr, thr, 0, stream>>>(Wq, Wk, Wv, Wt);
  }
  {
    const int rows = Bn * Tn;          // 16384
    const int blocks = rows / 64;      // 256 blocks x 4 waves x 16 rows
    sh_qkv_proj<<<blocks, 128, 0, stream>>>(x, bq, bk, bv, Wt, qbf, kbf, vT);
  }
  {
    const int blocks = Bn * (Tn / 64); // 256 blocks x 4 waves x 16-row tiles
    sh_flash_attn<<<blocks, 128, 0, stream>>>(qbf, kbf, vT, out);
  }
}
